// RelativeHead_30477087933203
// MI455X (gfx1250) — matrix operand order repaired and checked by reading, not compiled, not run on hardware
//
#include <hip/hip_runtime.h>
#include <hip/hip_bf16.h>

typedef _Float16 half_t;
typedef __attribute__((ext_vector_type(16))) _Float16 v16h;
typedef __attribute__((ext_vector_type(2)))  _Float16 h2;
typedef __attribute__((ext_vector_type(8)))  float    v8f;

#define T_SEQ  2048
#define NEMBD  768
#define HS     64
#define NBATCH 8
#define MAXREL (T_SEQ - 1)
#define SCALE  0.125f
#define LOG2E  1.4426950408889634f

__device__ __forceinline__ v8f wmma16(v16h a, v16h b, v8f c) {
  // D = A(16x32 f16) * B(32x16 f16) + C(16x16 f32)
  return __builtin_amdgcn_wmma_f32_16x16x32_f16(false, a, false, b, (short)0, c,
                                                false, false);
}

// ---------------------------------------------------------------------------
// Kernel 1: q = x @ Wq   (f32 in -> f16 out for WMMA consumption)
// 256 threads = 8 waves, 16 rows/wave -> 128 rows/block, 128 blocks.
// ---------------------------------------------------------------------------
__global__ __launch_bounds__(256) void proj_q_kernel(
    const float* __restrict__ x, const float* __restrict__ Wq,
    half_t* __restrict__ qh) {
  __shared__ __attribute__((aligned(16))) half_t wq_t[HS * 32];  // [col][kk]

  const int lane = threadIdx.x & 31;
  const int wave = threadIdx.x >> 5;
  const int r    = lane & 15;
  const int hi   = lane >> 4;                 // half-wave select
  const int rowBase = blockIdx.x * 128 + wave * 16;
  const float* xrow = x + (size_t)(rowBase + r) * NEMBD;
  const int kb = hi * 16;

  v8f c0 = {}, c1 = {}, c2 = {}, c3 = {};

  for (int k0 = 0; k0 < NEMBD; k0 += 32) {
    __syncthreads();
    // stage Wq[k0..k0+31, :] transposed so B-operand pairs are contiguous
    for (int i = threadIdx.x; i < 32 * HS; i += 256) {
      int kk = i >> 6, col = i & 63;
      wq_t[col * 32 + kk] = (half_t)Wq[(size_t)(k0 + kk) * HS + col];
    }
    __syncthreads();

    // A operand: 16x32 f16, contiguous k = 16*hi + s (convert f32->f16 on the fly)
    v16h a;
#pragma unroll
    for (int v = 0; v < 4; ++v) {
      int k = k0 + kb + 2 * v;
      a[2 * v]     = (half_t)xrow[k];
      a[2 * v + 1] = (half_t)xrow[k + 1];
      int k2 = k0 + 8 + kb + 2 * v;
      a[8 + 2 * v]     = (half_t)xrow[k2];
      a[8 + 2 * v + 1] = (half_t)xrow[k2 + 1];
    }

    const int kbB = hi * 16;
#pragma unroll
    for (int t = 0; t < 4; ++t) {
      const half_t* src = &wq_t[(t * 16 + r) * 32 + kbB];
      v16h b;
#pragma unroll
      for (int v = 0; v < 8; ++v) {
        h2 p = *(const h2*)(src + 2 * v);
        b[2 * v] = p.x; b[2 * v + 1] = p.y;
      }
      if      (t == 0) c0 = wmma16(a, b, c0);
      else if (t == 1) c1 = wmma16(a, b, c1);
      else if (t == 2) c2 = wmma16(a, b, c2);
      else             c3 = wmma16(a, b, c3);
    }
  }

  // C layout: VGPR j holds rows j (lanes 0-15) / j+8 (lanes 16-31), N = r
  half_t* qo = qh + (size_t)(rowBase + hi * 8) * HS;
#pragma unroll
  for (int j = 0; j < 8; ++j) {
    qo[(size_t)j * HS + r]      = (half_t)c0[j];
    qo[(size_t)j * HS + 16 + r] = (half_t)c1[j];
    qo[(size_t)j * HS + 32 + r] = (half_t)c2[j];
    qo[(size_t)j * HS + 48 + r] = (half_t)c3[j];
  }
}

// ---------------------------------------------------------------------------
// Kernel 2: flash attention with relative bias + causal mask (k = v = q).
// 256 threads = 8 waves; wave owns 16 query rows; block shares 32-key tiles.
// K/V tiles staged global->LDS with async copies (ASYNCcnt path) + prefetch.
// ---------------------------------------------------------------------------
__global__ __launch_bounds__(256) void rel_attn_kernel(
    const half_t* __restrict__ qh, const float* __restrict__ rel,
    float* __restrict__ out) {
  __shared__ __attribute__((aligned(16))) half_t kv[32 * HS];   // [key][h]
  __shared__ __attribute__((aligned(16))) half_t kvT[HS * 32];  // [h][key]
  __shared__ __attribute__((aligned(16))) half_t pst[8][16 * 32];

  const int lane = threadIdx.x & 31;
  const int wave = threadIdx.x >> 5;
  const int r    = lane & 15;
  const int hi   = lane >> 4;
  const int bpb  = T_SEQ / 128;                  // 16 blocks per batch
  const int b       = blockIdx.x / bpb;
  const int rowBase = (blockIdx.x % bpb) * 128;
  const int m0      = rowBase + wave * 16;
  const int kb      = hi * 16;

  const half_t* qbase = qh + (size_t)b * T_SEQ * HS;

  // this thread's 16-byte slice of the 4KB K/V tile (contiguous copy)
  const unsigned kv_lds = (unsigned)(uintptr_t)(&kv[0]) + threadIdx.x * 16u;
  const int e0 = threadIdx.x * 8;                // first half_t element index

  // Q tile in A layout: a_lo = head dims 0..31, a_hi = 32..63
  const half_t* qrow = qbase + (size_t)(m0 + r) * HS;
  v16h a_lo, a_hi;
#pragma unroll
  for (int v = 0; v < 4; ++v) {
    h2 p0 = *(const h2*)(qrow + kb + 2 * v);
    a_lo[2 * v] = p0.x; a_lo[2 * v + 1] = p0.y;
    h2 p1 = *(const h2*)(qrow + 8 + kb + 2 * v);
    a_lo[8 + 2 * v] = p1.x; a_lo[8 + 2 * v + 1] = p1.y;
    h2 p2 = *(const h2*)(qrow + 32 + kb + 2 * v);
    a_hi[2 * v] = p2.x; a_hi[2 * v + 1] = p2.y;
    h2 p3 = *(const h2*)(qrow + 40 + kb + 2 * v);
    a_hi[8 + 2 * v] = p3.x; a_hi[8 + 2 * v + 1] = p3.y;
  }

  v8f o0 = {}, o1 = {}, o2 = {}, o3 = {};
  float mrow[8], lrow[8];
#pragma unroll
  for (int j = 0; j < 8; ++j) { mrow[j] = -__builtin_inff(); lrow[j] = 0.0f; }

  const int nchunk = rowBase / 32 + 4;           // covers rows rowBase..+127
  for (int ch = 0; ch < nchunk; ++ch) {
    const int keybase = ch * 32;
    __syncthreads();                             // prev-tile readers done

    // async global->LDS copy of the 32-key tile (contiguous 4KB slab of qh)
    {
      unsigned long long ga =
          (unsigned long long)(uintptr_t)(qbase + (size_t)keybase * HS) +
          (unsigned long long)threadIdx.x * 16ull;
      asm volatile("global_load_async_to_lds_b128 %0, %1, off"
                   :: "v"(kv_lds), "v"(ga) : "memory");
    }
    // prefetch next chunk's tile into GL2 while this one lands
    if (ch + 1 < nchunk)
      __builtin_prefetch(qbase + (size_t)(keybase + 32) * HS + e0, 0, 0);

    asm volatile("s_wait_asynccnt 0" ::: "memory");
    // build transposed copy from this thread's own (landed) 16 bytes
    {
      const half_t* my = kv + e0;                // one ds_load_b128
#pragma unroll
      for (int j = 0; j < 8; ++j) {
        int e = e0 + j;
        kvT[(e & 63) * 32 + (e >> 6)] = my[j];
      }
    }
    __syncthreads();                             // tile ready for all waves
    if (keybase > m0 + 15) continue;             // wave-uniform causal skip

    // ---- QK^T: two 16-key subtiles, K-dim 64 split into 2 WMMA each ----
    float sv[2][8];
#pragma unroll
    for (int st = 0; st < 2; ++st) {
      const half_t* ks = kv + (st * 16 + r) * HS + hi * 16;
      v16h b0, b1;
#pragma unroll
      for (int v = 0; v < 8; ++v) {
        h2 p0 = *(const h2*)(ks + 2 * v);
        b0[2 * v] = p0.x; b0[2 * v + 1] = p0.y;
        h2 p1 = *(const h2*)(ks + 32 + 2 * v);
        b1[2 * v] = p1.x; b1[2 * v + 1] = p1.y;
      }
      v8f s = {};
      s = wmma16(a_lo, b0, s);
      s = wmma16(a_hi, b1, s);
      const int key = keybase + st * 16 + r;
#pragma unroll
      for (int j = 0; j < 8; ++j) {
        int row = m0 + hi * 8 + j;
        float val = s[j] * SCALE + rel[key - row + MAXREL];
        sv[st][j] = (key > row) ? -__builtin_inff() : val;
      }
    }

    // ---- online softmax: 16-lane half-wave row reductions ----
#pragma unroll
    for (int j = 0; j < 8; ++j) {
      float tmax = fmaxf(sv[0][j], sv[1][j]);
#pragma unroll
      for (int msk = 8; msk >= 1; msk >>= 1)
        tmax = fmaxf(tmax, __shfl_xor(tmax, msk, 32));
      float mnew  = fmaxf(mrow[j], tmax);
      float alpha = __builtin_exp2f((mrow[j] - mnew) * LOG2E);
      float p0 = __builtin_exp2f((sv[0][j] - mnew) * LOG2E);
      float p1 = __builtin_exp2f((sv[1][j] - mnew) * LOG2E);
      float ps = p0 + p1;
#pragma unroll
      for (int msk = 8; msk >= 1; msk >>= 1)
        ps += __shfl_xor(ps, msk, 32);
      lrow[j] = lrow[j] * alpha + ps;
      mrow[j] = mnew;
      o0[j] *= alpha; o1[j] *= alpha; o2[j] *= alpha; o3[j] *= alpha;
      // stage P (C layout -> LDS) for the register-layout transpose
      pst[wave][(hi * 8 + j) * 32 + r]      = (half_t)p0;
      pst[wave][(hi * 8 + j) * 32 + 16 + r] = (half_t)p1;
    }
    asm volatile("s_wait_dscnt 0" ::: "memory");

    // read P back in A layout (16x32 f16)
    const half_t* ps16 = &pst[wave][r * 32];
    v16h pA;
#pragma unroll
    for (int v = 0; v < 4; ++v) {
      h2 q0 = *(const h2*)(ps16 + kb + 2 * v);
      pA[2 * v] = q0.x; pA[2 * v + 1] = q0.y;
      h2 q1 = *(const h2*)(ps16 + 8 + kb + 2 * v);
      pA[8 + 2 * v] = q1.x; pA[8 + 2 * v + 1] = q1.y;
    }

    // ---- PV: P(16x32) @ V(32x64) as four N-tiles ----
#pragma unroll
    for (int t = 0; t < 4; ++t) {
      const half_t* vs = kvT + (t * 16 + r) * 32 + hi * 16;
      v16h bv;
#pragma unroll
      for (int v = 0; v < 8; ++v) {
        h2 p = *(const h2*)(vs + 2 * v);
        bv[2 * v] = p.x; bv[2 * v + 1] = p.y;
      }
      if      (t == 0) o0 = wmma16(pA, bv, o0);
      else if (t == 1) o1 = wmma16(pA, bv, o1);
      else if (t == 2) o2 = wmma16(pA, bv, o2);
      else             o3 = wmma16(pA, bv, o3);
    }
  }

  // epilogue: divide by softmax denominator, store f32
  float* ob = out + ((size_t)b * T_SEQ + m0 + hi * 8) * HS;
#pragma unroll
  for (int j = 0; j < 8; ++j) {
    float inv = 1.0f / lrow[j];
    ob[(size_t)j * HS + r]      = o0[j] * inv;
    ob[(size_t)j * HS + 16 + r] = o1[j] * inv;
    ob[(size_t)j * HS + 32 + r] = o2[j] * inv;
    ob[(size_t)j * HS + 48 + r] = o3[j] * inv;
  }
}

// ---------------------------------------------------------------------------
extern "C" void kernel_launch(void* const* d_in, const int* in_sizes, int n_in,
                              void* d_out, int out_size, void* d_ws,
                              size_t ws_size, hipStream_t stream) {
  (void)in_sizes; (void)n_in; (void)out_size; (void)ws_size;
  const float* x   = (const float*)d_in[0];   // [8, 2048, 768]
  const float* Wq  = (const float*)d_in[1];   // [768, 64]
  const float* rel = (const float*)d_in[2];   // [4095, 1]
  float* out = (float*)d_out;                 // [8, 2048, 64]
  half_t* qh = (half_t*)d_ws;                 // 16384*64 f16 = 2 MB scratch

  proj_q_kernel<<<128, 256, 0, stream>>>(x, Wq, qh);
  rel_attn_kernel<<<NBATCH * (T_SEQ / 128), 256, 0, stream>>>(qh, rel, out);
}
